// MultiScaleRetention_78082505441572
// MI455X (gfx1250) — compile-verified, matrix operand order checked by reading, not run on hardware
//
#include <hip/hip_runtime.h>

// ---------------------------------------------------------------------------
// MultiScaleRetention on MI455X (gfx1250, wave32)
//   two 25600x1024x1024 GEMMs -> bf16 WMMA (v_wmma_f32_16x16x32_bf16)
//   Mtile=32 per workgroup (512 thr, 16 waves = 2 Mgrp x 8 Ngrp) to halve
//   L2 weight re-reads (the binding resource per roofline), A staged in LDS
//   in two K-halves via GLOBAL_LOAD_ASYNC_TO_LDS_B128 (ASYNCcnt tracked).
// ---------------------------------------------------------------------------

typedef __attribute__((ext_vector_type(16))) __bf16 v16bf;
typedef __attribute__((ext_vector_type(8)))  float  v8f;
typedef __attribute__((ext_vector_type(4)))  int    v4i;

union Frag {            // 16 bf16 = 8 VGPRs, loaded as two 16B chunks
    v16bf bf;
    uint4 q[2];
};

__device__ __forceinline__ unsigned short f2bf(float f) {
    union { float f; unsigned u; } c; c.f = f;
    unsigned u = c.u;
    u += 0x7FFFu + ((u >> 16) & 1u);        // round-to-nearest-even
    return (unsigned short)(u >> 16);
}

#define SEQL   50
#define DIMM   1024
#define NHEAD  32
#define BSZ    512
#define EPSV   1e-5f
#define KHALF  512
#define KPITCH 520    // LDS row pitch (elements): 512 + 8 pad -> 4-bank stride

// ---- async global->LDS staging (CDNA5 path, guarded with fallback) --------
#if defined(__gfx1250__) && __has_builtin(__builtin_amdgcn_global_load_async_to_lds_b128)
#define HAS_ASYNC_LDS 1
#else
#define HAS_ASYNC_LDS 0
#endif

__device__ __forceinline__ void async_copy16(const unsigned short* g, unsigned short* l) {
#if HAS_ASYNC_LDS
    unsigned short* gg = (unsigned short*)g;      // builtin takes non-const
    __builtin_amdgcn_global_load_async_to_lds_b128(
        (__attribute__((address_space(1))) v4i*)gg,
        (__attribute__((address_space(3))) v4i*)l, 0, 0);
#else
    *(uint4*)l = *(const uint4*)g;
#endif
}

__device__ __forceinline__ void async_join() {
#if HAS_ASYNC_LDS && __has_builtin(__builtin_amdgcn_s_wait_asynccnt)
    __builtin_amdgcn_s_wait_asynccnt(0);
#endif
    __syncthreads();
}

// ---------------------------------------------------------------------------
// fp32 -> bf16 conversion (grid-stride)
// ---------------------------------------------------------------------------
__global__ void msr_f32_to_bf16(const float* __restrict__ in,
                                unsigned short* __restrict__ out, long n) {
    long i = (long)blockIdx.x * blockDim.x + threadIdx.x;
    long s = (long)gridDim.x * blockDim.x;
    for (; i < n; i += s) out[i] = f2bf(in[i]);
}

// ---------------------------------------------------------------------------
// RNN scan: h_{b+1} = tanh(c + whh h_b + bhh), c = gamma*sum_s wih[:,s] + bih
// One head per workgroup; whh row in VGPRs, state in LDS; 512 serial steps.
// S stored (B, H, SEQ).
// ---------------------------------------------------------------------------
__global__ __launch_bounds__(64)
void msr_rnn(const float* __restrict__ wih, const float* __restrict__ whh,
             const float* __restrict__ bih, const float* __restrict__ bhh,
             float* __restrict__ S) {
    __shared__ float hbuf[SEQL];
    const int h = blockIdx.x;
    const int o = threadIdx.x;
    float wrow[SEQL];
    float c = 0.f;
    if (o < SEQL) {
        const float* wi = wih + ((size_t)h * SEQL + o) * SEQL;
        const float* wh = whh + ((size_t)h * SEQL + o) * SEQL;
        float s = 0.f;
        #pragma unroll
        for (int i = 0; i < SEQL; ++i) { s += wi[i]; wrow[i] = wh[i]; }
        float gamma = 1.f - exp2f(-(float)(5 + h));
        c = gamma * s + bih[h * SEQL + o] + bhh[h * SEQL + o];
        hbuf[o] = 0.f;
    }
    __syncthreads();
    for (int b = 0; b < BSZ; ++b) {
        float val = 0.f;
        if (o < SEQL) {
            float a = c;
            #pragma unroll
            for (int i = 0; i < SEQL; ++i) a += hbuf[i] * wrow[i];
            val = tanhf(a);
        }
        __syncthreads();
        if (o < SEQL) {
            hbuf[o] = val;
            S[((size_t)b * NHEAD + h) * SEQL + o] = val;
        }
        __syncthreads();
    }
}

// ---------------------------------------------------------------------------
// Fused kernel 1: per workgroup 32 rows x all 1024 cols.
//   gate = x @ wg^T (bf16 WMMA, K=1024 in two staged K-halves)
//   Q    = per-head x @ wq^T (one WMMA per 16-col tile, K=32, in its phase)
//   Y0   = S*Q ; GroupNorm over DIM ; Z = silu(gate)*Ynorm (bf16)
// ---------------------------------------------------------------------------
__global__ __launch_bounds__(512)
void msr_gate_q_gn(const unsigned short* __restrict__ xb,
                   const unsigned short* __restrict__ wgb,
                   const unsigned short* __restrict__ wqb,
                   const float* __restrict__ S,
                   const float* __restrict__ gnw,
                   const float* __restrict__ gnb,
                   unsigned short* __restrict__ Zb) {
    __shared__ __align__(16) unsigned short At[32 * KPITCH];
    __shared__ float Srow[32 * NHEAD];
    __shared__ float rsum[32];
    __shared__ float rsq[32];

    const int tid  = threadIdx.x;
    const int wave = tid >> 5;
    const int lane = tid & 31;
    const int m16  = lane & 15;    // A row / B,D column within 16x16 tile
    const int kh   = lane >> 4;    // K-half selector inside fragment
    const int hi   = kh;           // D rows m = v + 8*hi
    const int wm   = wave >> 3;    // M-group (0..1): rows wm*16 .. wm*16+15
    const int wn   = wave & 7;     // N-group (0..7): cols wn*128 .. +127
    const int r0   = blockIdx.x * 32;

    // stage S for these 32 rows x 32 heads; init stats accumulators
    #pragma unroll
    for (int it = 0; it < 2; ++it) {
        int idx = tid * 2 + it;                       // 0..1023
        int row = idx >> 5, h = idx & 31;
        int r = r0 + row, b = r / SEQL, t = r - b * SEQL;
        Srow[row * NHEAD + h] = S[((size_t)b * NHEAD + h) * SEQL + t];
    }
    if (tid < 32) { rsum[tid] = 0.f; rsq[tid] = 0.f; }

    const int nbase = wn * 128;
    const unsigned short* bp[8];
    #pragma unroll
    for (int t = 0; t < 8; ++t) {
        int e = nbase + t * 16 + m16;                 // B column = wg row e
        bp[t] = wgb + (size_t)e * DIMM + kh * 8;
    }
    const unsigned short* ap = &At[(wm * 16 + m16) * KPITCH + kh * 8];

    v8f acc[8] = {};
    v8f qacc[8];
    for (int p = 0; p < 2; ++p) {
        if (p) __syncthreads();                       // phase-0 readers done
        // stage 32 rows x 512 bf16 of A for this K-half (2048 x 16B chunks)
        #pragma unroll
        for (int it = 0; it < 4; ++it) {
            int idx = tid + it * 512;
            int row = idx >> 6, c = idx & 63;
            async_copy16(xb + (size_t)(r0 + row) * DIMM + p * KHALF + c * 8,
                         &At[row * KPITCH + c * 8]);
        }
        async_join();                                 // s_wait_asynccnt + barrier

        for (int k0 = 0; k0 < KHALF; k0 += 32) {
            Frag a;
            a.q[0] = *(const uint4*)(ap + k0);
            a.q[1] = *(const uint4*)(ap + k0 + 16);
            int kg = p * KHALF + k0;
            #pragma unroll
            for (int t = 0; t < 8; ++t) {
                Frag b;
                b.q[0] = *(const uint4*)(bp[t] + kg);
                b.q[1] = *(const uint4*)(bp[t] + kg + 16);
                acc[t] = __builtin_amdgcn_wmma_f32_16x16x32_bf16(
                             false, a.bf, false, b.bf, (short)0, acc[t], false, false);
            }
        }

        // Q tiles: head h = e0/32; its K=32 slice lives in phase (wn>>2)
        if ((wn >> 2) == p) {
            #pragma unroll
            for (int t = 0; t < 8; ++t) {
                int e0 = nbase + t * 16;
                int h  = e0 >> 5;
                int kl = h * 32 - p * KHALF;          // local offset in phase
                Frag a, b;
                a.q[0] = *(const uint4*)(ap + kl);
                a.q[1] = *(const uint4*)(ap + kl + 16);
                const unsigned short* qp =
                    wqb + ((size_t)h * 32 + (e0 & 31) + m16) * 32 + kh * 8;
                b.q[0] = *(const uint4*)(qp);
                b.q[1] = *(const uint4*)(qp + 16);
                v8f z = {};
                qacc[t] = __builtin_amdgcn_wmma_f32_16x16x32_bf16(
                              false, a.bf, false, b.bf, (short)0, z, false, false);
            }
        }
    }

    // row stats of Y0 = S*Q over all 1024 columns
    float sy[8] = {}, syy[8] = {};
    #pragma unroll
    for (int t = 0; t < 8; ++t) {
        int h = (nbase + t * 16) >> 5;
        #pragma unroll
        for (int v = 0; v < 8; ++v) {
            float y = Srow[(wm * 16 + v + 8 * hi) * NHEAD + h] * qacc[t][v];
            sy[v] += y; syy[v] += y * y;
        }
    }
    #pragma unroll
    for (int v = 0; v < 8; ++v)
        #pragma unroll
        for (int off = 1; off < 16; off <<= 1) {
            sy[v]  += __shfl_xor(sy[v],  off, 16);
            syy[v] += __shfl_xor(syy[v], off, 16);
        }
    if (m16 == 0)
        #pragma unroll
        for (int v = 0; v < 8; ++v) {
            atomicAdd(&rsum[wm * 16 + v + 8 * hi], sy[v]);   // ds_add_f32
            atomicAdd(&rsq[wm * 16 + v + 8 * hi],  syy[v]);
        }
    __syncthreads();

    // epilogue: GroupNorm (group size 1 over DIM) + SiLU(gate)*Ynorm -> bf16
    #pragma unroll
    for (int v = 0; v < 8; ++v) {
        int mr = wm * 16 + v + 8 * hi;
        int r  = r0 + mr;
        int b  = r / SEQL, ts = r - b * SEQL;
        float mu   = rsum[mr] * (1.f / DIMM);
        float var  = rsq[mr]  * (1.f / DIMM) - mu * mu;
        float rstd = rsqrtf(var + EPSV);
        float gw = gnw[ts], gb = gnb[ts];
        size_t orow = (size_t)r * DIMM;
        #pragma unroll
        for (int t = 0; t < 8; ++t) {
            int e = nbase + t * 16 + m16;
            int h = (nbase + t * 16) >> 5;
            float g  = acc[t][v];
            float sg = g / (1.f + __expf(-g));                 // SiLU
            float y  = Srow[mr * NHEAD + h] * qacc[t][v];
            float yn = (y - mu) * rstd * gw + gb;
            Zb[orow + e] = f2bf(sg * yn);
        }
    }
}

// ---------------------------------------------------------------------------
// Kernel 2: out = Z @ wo^T  (bf16 WMMA, fp32 output), Mtile=32
// ---------------------------------------------------------------------------
__global__ __launch_bounds__(512)
void msr_out_gemm(const unsigned short* __restrict__ Zb,
                  const unsigned short* __restrict__ wob,
                  float* __restrict__ out) {
    __shared__ __align__(16) unsigned short At[32 * KPITCH];

    const int tid  = threadIdx.x;
    const int wave = tid >> 5;
    const int lane = tid & 31;
    const int m16  = lane & 15;
    const int kh   = lane >> 4;
    const int hi   = kh;
    const int wm   = wave >> 3;
    const int wn   = wave & 7;
    const int r0   = blockIdx.x * 32;

    const int nbase = wn * 128;
    const unsigned short* bp[8];
    #pragma unroll
    for (int t = 0; t < 8; ++t) {
        int e = nbase + t * 16 + m16;
        bp[t] = wob + (size_t)e * DIMM + kh * 8;
    }
    const unsigned short* ap = &At[(wm * 16 + m16) * KPITCH + kh * 8];

    v8f acc[8] = {};
    for (int p = 0; p < 2; ++p) {
        if (p) __syncthreads();
        #pragma unroll
        for (int it = 0; it < 4; ++it) {
            int idx = tid + it * 512;
            int row = idx >> 6, c = idx & 63;
            async_copy16(Zb + (size_t)(r0 + row) * DIMM + p * KHALF + c * 8,
                         &At[row * KPITCH + c * 8]);
        }
        async_join();

        for (int k0 = 0; k0 < KHALF; k0 += 32) {
            Frag a;
            a.q[0] = *(const uint4*)(ap + k0);
            a.q[1] = *(const uint4*)(ap + k0 + 16);
            int kg = p * KHALF + k0;
            #pragma unroll
            for (int t = 0; t < 8; ++t) {
                Frag b;
                b.q[0] = *(const uint4*)(bp[t] + kg);
                b.q[1] = *(const uint4*)(bp[t] + kg + 16);
                acc[t] = __builtin_amdgcn_wmma_f32_16x16x32_bf16(
                             false, a.bf, false, b.bf, (short)0, acc[t], false, false);
            }
        }
    }

    #pragma unroll
    for (int v = 0; v < 8; ++v) {
        size_t orow = (size_t)(r0 + wm * 16 + v + 8 * hi) * DIMM;
        #pragma unroll
        for (int t = 0; t < 8; ++t)
            out[orow + nbase + t * 16 + m16] = acc[t][v];
    }
}

// ---------------------------------------------------------------------------
// Launch
// ---------------------------------------------------------------------------
extern "C" void kernel_launch(void* const* d_in, const int* in_sizes, int n_in,
                              void* d_out, int out_size, void* d_ws, size_t ws_size,
                              hipStream_t stream) {
    (void)in_sizes; (void)n_in; (void)out_size; (void)ws_size;
    const float* x   = (const float*)d_in[0];
    const float* wq  = (const float*)d_in[1];
    // d_in[2] = wk, d_in[3] = wv : dead code in the reference (DCE'd)
    const float* wih = (const float*)d_in[4];
    const float* whh = (const float*)d_in[5];
    const float* bih = (const float*)d_in[6];
    const float* bhh = (const float*)d_in[7];
    const float* gnw = (const float*)d_in[8];
    const float* gnb = (const float*)d_in[9];
    const float* wg  = (const float*)d_in[10];
    const float* wo  = (const float*)d_in[11];
    float* out = (float*)d_out;

    const long NX = (long)BSZ * SEQL * DIMM;    // 26,214,400
    const long NW = (long)DIMM * DIMM;          // 1,048,576
    const long NQ = (long)NHEAD * 32 * 32;      // 32,768

    char* ws = (char*)d_ws;
    unsigned short* xb  = (unsigned short*)ws;  ws += NX * 2;
    unsigned short* zb  = (unsigned short*)ws;  ws += NX * 2;
    unsigned short* wgb = (unsigned short*)ws;  ws += NW * 2;
    unsigned short* wob = (unsigned short*)ws;  ws += NW * 2;
    unsigned short* wqb = (unsigned short*)ws;  ws += NQ * 2;
    float*          Sbuf = (float*)ws;          ws += (long)BSZ * NHEAD * SEQL * 4;

    msr_f32_to_bf16<<<2048, 256, 0, stream>>>(x,  xb,  NX);
    msr_f32_to_bf16<<<256,  256, 0, stream>>>(wg, wgb, NW);
    msr_f32_to_bf16<<<256,  256, 0, stream>>>(wo, wob, NW);
    msr_f32_to_bf16<<<32,   256, 0, stream>>>(wq, wqb, NQ);

    msr_rnn<<<32, 64, 0, stream>>>(wih, whh, bih, bhh, Sbuf);

    msr_gate_q_gn<<<800, 512, 0, stream>>>(xb, wgb, wqb, Sbuf, gnw, gnb, zb);
    msr_out_gemm <<<800, 512, 0, stream>>>(zb, wob, out);
}
